// PicktEncoderSelfAttention_3745211482569
// MI455X (gfx1250) — matrix operand order repaired and checked by reading, not compiled, not run on hardware
//
#include <hip/hip_runtime.h>
#include <hip/hip_bf16.h>
#include <math.h>

// ---------------- types ----------------
typedef __bf16 bf16;
typedef __attribute__((ext_vector_type(8)))  float v8f;
typedef __attribute__((ext_vector_type(8)))  bf16  v8bf;
typedef __attribute__((ext_vector_type(16))) bf16  v16bf;

// ---------------- problem sizes ----------------
#define BB   4
#define SS   1024
#define EE   768
#define HH   12
#define DD   64
#define NTOK (BB*SS)          // 4096
#define BSE  (BB*SS*EE)       // 3145728
#define NDE  2048             // 2047 real rows + 1 zero pad

// ---------------- workspace layout (bytes) ----------------
#define OFF_XH   ((size_t)0)
#define OFF_XC   (OFF_XH  + (size_t)NTOK*EE*2)
#define OFF_WT   (OFF_XC  + (size_t)NTOK*EE*2)
#define OFF_DE   (OFF_WT  + (size_t)6*EE*EE*2)
#define OFF_PQQ  (OFF_DE  + (size_t)NDE*DD*2)
#define OFF_PQK  (OFF_PQQ + (size_t)NTOK*EE*2)
#define OFF_PCQ  (OFF_PQK + (size_t)NTOK*EE*2)
#define OFF_PCK  (OFF_PCQ + (size_t)NTOK*EE*2)
#define OFF_VTQ  (OFF_PCK + (size_t)NTOK*EE*2)
#define OFF_VTC  (OFF_VTQ + (size_t)BB*HH*DD*SS*2)

// ---------------- WMMA helpers ----------------
__device__ inline v8f zero8() {
  v8f r;
#pragma unroll
  for (int i = 0; i < 8; ++i) r[i] = 0.f;
  return r;
}

__device__ inline v16bf frag_combine(v8bf lo, v8bf hi) {
  v16bf f;
#pragma unroll
  for (int i = 0; i < 8; ++i) { f[i] = lo[i]; f[i + 8] = hi[i]; }
  return f;
}

// A-matrix fragment 16x32 bf16 (ISA 7.12.2):
// lanes 0-15 : row = lane,    K chunks {0..7, 16..23}
// lanes 16-31: row = lane-16, K chunks {8..15, 24..31}
__device__ inline v16bf load_fragA(const bf16* base, int ld, int lane) {
  int r = lane & 15, g = (lane >> 4) & 1;
  const bf16* p = base + (size_t)r * ld + g * 8;
  v8bf lo = *(const v8bf*)p;
  v8bf hi = *(const v8bf*)(p + 16);
  return frag_combine(lo, hi);
}

// B-matrix fragment 32x16 bf16: col n = lane&15,
// lanes 0-15 hold K chunks {0..7, 16..23}, lanes 16-31 hold K {8..15, 24..31}
__device__ inline v16bf load_fragB(const bf16* base, int ld, int lane) {
  int n = lane & 15, g = (lane >> 4) & 1;
  const bf16* p = base + (size_t)n * ld + g * 8;
  v8bf lo = *(const v8bf*)p;
  v8bf hi = *(const v8bf*)(p + 16);
  return frag_combine(lo, hi);
}

__device__ inline v8f wmma_bf(v16bf a, v16bf b, v8f c) {
  return __builtin_amdgcn_wmma_f32_16x16x32_bf16(false, a, false, b, (short)0, c,
                                                 false, false);
}

// ---------------- conversion kernels ----------------
__global__ void cvt_f32_bf16(const float* __restrict__ src, bf16* __restrict__ dst,
                             int n) {
  for (int i = blockIdx.x * blockDim.x + threadIdx.x; i < n;
       i += gridDim.x * blockDim.x)
    dst[i] = (bf16)src[i];
}

__global__ void cvt_dist_emb(const float* __restrict__ src, bf16* __restrict__ dst) {
  int i = blockIdx.x * blockDim.x + threadIdx.x;   // total NDE*DD = 131072
  if (i >= NDE * DD) return;
  dst[i] = (i < 2047 * DD) ? (bf16)src[i] : (bf16)0.f;
}

// transpose + convert six [768,768] weights: Wt[p][n][k] = W_p[k][n]
__global__ void wtrans_kernel(const float* w0, const float* w1, const float* w2,
                              const float* w3, const float* w4, const float* w5,
                              bf16* __restrict__ Wt) {
  __shared__ float t[32][33];
  int p = blockIdx.z;
  const float* W = (p == 0) ? w0 : (p == 1) ? w1 : (p == 2) ? w2
                 : (p == 3) ? w3 : (p == 4) ? w4 : w5;
  bf16* dst = Wt + (size_t)p * EE * EE;
  int n0 = blockIdx.x * 32, k0 = blockIdx.y * 32;
  int tx = threadIdx.x, ty = threadIdx.y;       // 32 x 8
#pragma unroll
  for (int i = 0; i < 4; ++i)
    t[ty + 8 * i][tx] = W[(size_t)(k0 + ty + 8 * i) * EE + n0 + tx];
  __syncthreads();
#pragma unroll
  for (int i = 0; i < 4; ++i)
    dst[(size_t)(n0 + ty + 8 * i) * EE + k0 + tx] = (bf16)t[tx][ty + 8 * i];
}

// ---------------- projection GEMM: P = X @ W + b ----------------
// grid (6, 32, 6) : x = N/128, y = M/128, z = projection
// block 256 = 8 waves; wave: 64x32 region (4 M-tiles x 2 N-tiles)
__global__ __launch_bounds__(256) void proj_gemm(
    const bf16* __restrict__ Xh, const bf16* __restrict__ Xc,
    const bf16* __restrict__ Wt,
    const float* b0, const float* b1, const float* b2, const float* b3,
    const float* b4, const float* b5,
    bf16* Pqq, bf16* Pqk, bf16* Pcq, bf16* Pck, bf16* Vtq, bf16* Vtc) {
  int p = blockIdx.z;
  const bf16* X = (p < 3) ? Xh : Xc;
  const bf16* W = Wt + (size_t)p * EE * EE;
  const float* bias = (p == 0) ? b0 : (p == 1) ? b1 : (p == 2) ? b2
                    : (p == 3) ? b3 : (p == 4) ? b4 : b5;
  bf16* dst; bool tr = false;
  switch (p) {
    case 0: dst = Pqq; break;
    case 1: dst = Pqk; break;
    case 2: dst = Vtq; tr = true; break;
    case 3: dst = Pcq; break;
    case 4: dst = Pck; break;
    default: dst = Vtc; tr = true; break;
  }
  int tid = threadIdx.x, lane = tid & 31, w = tid >> 5;
  int wm = w & 1, wn = w >> 1;                       // 2 x 4 wave grid
  int m0 = blockIdx.y * 128 + wm * 64;
  int n0 = blockIdx.x * 128 + wn * 32;
  int row16 = lane & 15, g = (lane >> 4) & 1;

  v8f acc[4][2];
#pragma unroll
  for (int mt = 0; mt < 4; ++mt)
#pragma unroll
    for (int nt = 0; nt < 2; ++nt) acc[mt][nt] = zero8();

  for (int k0 = 0; k0 < EE; k0 += 32) {
    // prefetch next K-slab into the near (WGP) cache: locality 3
    if (k0 + 32 < EE) {
      __builtin_prefetch(X + (size_t)(m0 + row16 * 4) * EE + k0 + 32, 0, 3);
      __builtin_prefetch(W + (size_t)(n0 + row16 * 2) * EE + k0 + 32, 0, 3);
    }
    v16bf aF[4], bF[2];
#pragma unroll
    for (int mt = 0; mt < 4; ++mt)
      aF[mt] = load_fragA(X + (size_t)(m0 + mt * 16) * EE + k0, EE, lane);
#pragma unroll
    for (int nt = 0; nt < 2; ++nt)
      bF[nt] = load_fragB(W + (size_t)(n0 + nt * 16) * EE + k0, EE, lane);
#pragma unroll
    for (int mt = 0; mt < 4; ++mt)
#pragma unroll
      for (int nt = 0; nt < 2; ++nt)
        acc[mt][nt] = wmma_bf(aF[mt], bF[nt], acc[mt][nt]);
  }

  // epilogue: 'tr' is wave-uniform -> single branch, no per-store control flow
  if (!tr) {
#pragma unroll
    for (int nt = 0; nt < 2; ++nt) {
      int col = n0 + nt * 16 + row16;
      float bv = bias[col];
#pragma unroll
      for (int mt = 0; mt < 4; ++mt)
#pragma unroll
        for (int v = 0; v < 8; ++v) {
          int m = m0 + mt * 16 + v + 8 * g;          // token index
          dst[(size_t)m * EE + col] = (bf16)(acc[mt][nt][v] + bv);
        }
    }
  } else {
#pragma unroll
    for (int nt = 0; nt < 2; ++nt) {
      int col = n0 + nt * 16 + row16;
      float bv = bias[col];
      int hh = col >> 6, d = col & 63;
#pragma unroll
      for (int mt = 0; mt < 4; ++mt)
#pragma unroll
        for (int v = 0; v < 8; ++v) {
          int m = m0 + mt * 16 + v + 8 * g;
          int bb = m >> 10, s = m & 1023;            // [bh][d][s] layout
          dst[((size_t)((bb * HH + hh) * DD + d)) * SS + s] =
              (bf16)(acc[mt][nt][v] + bv);
        }
    }
  }
}

// ---------------- fused dual-stream attention (flash style) ----------------
// grid (S/16, B*H); one wave per workgroup; 16 queries, loop 32 keys/iter
__global__ __launch_bounds__(32) void attn_kernel(
    const bf16* __restrict__ Pqq, const bf16* __restrict__ Pqk,
    const bf16* __restrict__ Pcq, const bf16* __restrict__ Pck,
    const bf16* __restrict__ Vtq, const bf16* __restrict__ Vtc,
    const bf16* __restrict__ DE, float* __restrict__ out) {
  const int l0 = blockIdx.x * 16;
  const int bh = blockIdx.y;
  const int b = bh / HH, h = bh % HH;
  const int lane = threadIdx.x & 31;
  const int row16 = lane & 15, g = (lane >> 4) & 1;
  const float scale = 0.125f;                        // 1/sqrt(64)

  __shared__ float s_gq[16 * 48];
  __shared__ float s_gk[32 * 48];
  __shared__ __align__(16) bf16 s_pT[16 * 32];

  const bf16* qbase = Pqq + (size_t)(b * SS + l0) * EE + h * DD;
  const bf16* cbase = Pcq + (size_t)(b * SS + l0) * EE + h * DD;
  v16bf qqA[2], cqA[2];
#pragma unroll
  for (int sl = 0; sl < 2; ++sl) {
    qqA[sl] = load_fragA(qbase + sl * 32, EE, lane);
    cqA[sl] = load_fragA(cbase + sl * 32, EE, lane);
  }

  v8f accQ[4], accC[4];
#pragma unroll
  for (int dt = 0; dt < 4; ++dt) { accQ[dt] = zero8(); accC[dt] = zero8(); }
  float mrow[8], lrow[8];
#pragma unroll
  for (int v = 0; v < 8; ++v) { mrow[v] = -3.0e38f; lrow[v] = 0.f; }

  for (int r0 = 0; r0 < SS; r0 += 32) {
    // prefetch next key/value tiles into the near (WGP) cache
    if (r0 + 32 < SS) {
      const size_t nk = (size_t)(b * SS + r0 + 32 + row16 * 2) * EE + h * DD;
      __builtin_prefetch(Pqk + nk, 0, 3);
      __builtin_prefetch(Pck + nk, 0, 3);
      const size_t nv = (size_t)(bh * DD + row16 * 4) * SS + r0 + 32;
      __builtin_prefetch(Vtq + nv, 0, 3);
      __builtin_prefetch(Vtc + nv, 0, 3);
    }

    const bf16* kqb = Pqk + (size_t)(b * SS + r0) * EE + h * DD;
    const bf16* kcb = Pck + (size_t)(b * SS + r0) * EE + h * DD;
    v16bf qkB[2][2], ckB[2][2], qkA[2][2];
#pragma unroll
    for (int kt = 0; kt < 2; ++kt)
#pragma unroll
      for (int sl = 0; sl < 2; ++sl) {
        qkB[kt][sl] = load_fragB(kqb + (size_t)kt * 16 * EE + sl * 32, EE, lane);
        ckB[kt][sl] = load_fragB(kcb + (size_t)kt * 16 * EE + sl * 32, EE, lane);
        qkA[kt][sl] = load_fragA(kqb + (size_t)kt * 16 * EE + sl * 32, EE, lane);
      }

    // raw scores: qq.qk + cq.ck
    v8f sAcc[2];
#pragma unroll
    for (int kt = 0; kt < 2; ++kt) {
      v8f s = zero8();
      s = wmma_bf(qqA[0], qkB[kt][0], s);
      s = wmma_bf(qqA[1], qkB[kt][1], s);
      s = wmma_bf(cqA[0], ckB[kt][0], s);
      s = wmma_bf(cqA[1], ckB[kt][1], s);
      sAcc[kt] = s;
    }

    // relative-position pieces via WMMA against dist_emb slice
    int M0 = l0 - r0 + 992;                          // in [0, 2000]
#pragma unroll
    for (int t = 0; t < 3; ++t) {
      v16bf deB[2];
#pragma unroll
      for (int sl = 0; sl < 2; ++sl)
        deB[sl] = load_fragB(DE + (size_t)(M0 + t * 16) * DD + sl * 32, DD, lane);
      v8f gq = zero8();
      gq = wmma_bf(qqA[0], deB[0], gq);
      gq = wmma_bf(qqA[1], deB[1], gq);
#pragma unroll
      for (int v = 0; v < 8; ++v)
        s_gq[(v + 8 * g) * 48 + t * 16 + row16] = gq[v];
#pragma unroll
      for (int kt = 0; kt < 2; ++kt) {
        v8f gk = zero8();
        gk = wmma_bf(qkA[kt][0], deB[0], gk);
        gk = wmma_bf(qkA[kt][1], deB[1], gk);
#pragma unroll
        for (int v = 0; v < 8; ++v)
          s_gk[(kt * 16 + v + 8 * g) * 48 + t * 16 + row16] = gk[v];
      }
    }
    __syncthreads();   // 1-wave WG: waits + NOP barrier

    // gather positional terms, scale, online softmax
    float pv[2][8], rmax[8];
#pragma unroll
    for (int v = 0; v < 8; ++v) {
      int ll = v + 8 * g;
      float sv[2];
#pragma unroll
      for (int kt = 0; kt < 2; ++kt) {
        int rl = kt * 16 + row16;
        int j = 31 + ll - rl;                        // in [0,46]
        sv[kt] = (sAcc[kt][v] + s_gq[ll * 48 + j] + s_gk[rl * 48 + j]) * scale;
      }
      pv[0][v] = sv[0]; pv[1][v] = sv[1];
      rmax[v] = fmaxf(sv[0], sv[1]);
    }
#pragma unroll
    for (int v = 0; v < 8; ++v)
#pragma unroll
      for (int msk = 1; msk < 16; msk <<= 1)
        rmax[v] = fmaxf(rmax[v], __shfl_xor(rmax[v], msk, 32));

    float corr[8];
#pragma unroll
    for (int v = 0; v < 8; ++v) {
      float mn = fmaxf(mrow[v], rmax[v]);
      corr[v] = __expf(mrow[v] - mn);
      mrow[v] = mn;
      float e0 = __expf(pv[0][v] - mn);
      float e1 = __expf(pv[1][v] - mn);
      pv[0][v] = e0; pv[1][v] = e1;
      float rs = e0 + e1;
#pragma unroll
      for (int msk = 1; msk < 16; msk <<= 1) rs += __shfl_xor(rs, msk, 32);
      lrow[v] = lrow[v] * corr[v] + rs;
    }
#pragma unroll
    for (int dt = 0; dt < 4; ++dt)
#pragma unroll
      for (int v = 0; v < 8; ++v) {
        accQ[dt][v] *= corr[v];
        accC[dt][v] *= corr[v];
      }

    // probabilities -> LDS (bf16) -> A fragment for P @ V
    __syncthreads();
#pragma unroll
    for (int v = 0; v < 8; ++v) {
      int ll = v + 8 * g;
      s_pT[ll * 32 + row16]      = (bf16)pv[0][v];
      s_pT[ll * 32 + 16 + row16] = (bf16)pv[1][v];
    }
    __syncthreads();
    v16bf pA = load_fragA(s_pT, 32, lane);

#pragma unroll
    for (int dt = 0; dt < 4; ++dt) {
      v16bf vqB = load_fragB(Vtq + ((size_t)(bh * DD + dt * 16)) * SS + r0, SS, lane);
      v16bf vcB = load_fragB(Vtc + ((size_t)(bh * DD + dt * 16)) * SS + r0, SS, lane);
      accQ[dt] = wmma_bf(pA, vqB, accQ[dt]);
      accC[dt] = wmma_bf(pA, vcB, accC[dt]);
    }
    __syncthreads();
  }

  // epilogue: normalize and store fp32 outputs [2,B,S,E]
  float inv[8];
#pragma unroll
  for (int v = 0; v < 8; ++v) inv[v] = 1.0f / lrow[v];
#pragma unroll
  for (int dt = 0; dt < 4; ++dt) {
#pragma unroll
    for (int v = 0; v < 8; ++v) {
      int l = l0 + v + 8 * g;
      int col = h * DD + dt * 16 + row16;
      size_t o = (size_t)(b * SS + l) * EE + col;
      out[o]               = accQ[dt][v] * inv[v];
      out[(size_t)BSE + o] = accC[dt][v] * inv[v];
    }
  }
}

// ---------------- host-side launch ----------------
extern "C" void kernel_launch(void* const* d_in, const int* in_sizes, int n_in,
                              void* d_out, int out_size, void* d_ws, size_t ws_size,
                              hipStream_t stream) {
  (void)in_sizes; (void)n_in; (void)out_size; (void)ws_size;
  const float* hidden  = (const float*)d_in[0];
  const float* concept = (const float*)d_in[1];
  const float* w[6]    = {(const float*)d_in[2],  (const float*)d_in[4],
                          (const float*)d_in[6],  (const float*)d_in[8],
                          (const float*)d_in[10], (const float*)d_in[12]};
  const float* bias[6] = {(const float*)d_in[3],  (const float*)d_in[5],
                          (const float*)d_in[7],  (const float*)d_in[9],
                          (const float*)d_in[11], (const float*)d_in[13]};
  const float* dist = (const float*)d_in[14];
  float* out = (float*)d_out;

  char* ws = (char*)d_ws;
  bf16* Xh  = (bf16*)(ws + OFF_XH);
  bf16* Xc  = (bf16*)(ws + OFF_XC);
  bf16* Wt  = (bf16*)(ws + OFF_WT);
  bf16* DE  = (bf16*)(ws + OFF_DE);
  bf16* Pqq = (bf16*)(ws + OFF_PQQ);
  bf16* Pqk = (bf16*)(ws + OFF_PQK);
  bf16* Pcq = (bf16*)(ws + OFF_PCQ);
  bf16* Pck = (bf16*)(ws + OFF_PCK);
  bf16* Vtq = (bf16*)(ws + OFF_VTQ);
  bf16* Vtc = (bf16*)(ws + OFF_VTC);

  cvt_f32_bf16<<<2048, 256, 0, stream>>>(hidden, Xh, NTOK * EE);
  cvt_f32_bf16<<<2048, 256, 0, stream>>>(concept, Xc, NTOK * EE);
  wtrans_kernel<<<dim3(24, 24, 6), dim3(32, 8), 0, stream>>>(
      w[0], w[1], w[2], w[3], w[4], w[5], Wt);
  cvt_dist_emb<<<(NDE * DD + 255) / 256, 256, 0, stream>>>(dist, DE);

  proj_gemm<<<dim3(6, 32, 6), 256, 0, stream>>>(
      Xh, Xc, Wt, bias[0], bias[1], bias[2], bias[3], bias[4], bias[5],
      Pqq, Pqk, Pcq, Pck, Vtq, Vtc);

  attn_kernel<<<dim3(SS / 16, BB * HH), 32, 0, stream>>>(
      Pqq, Pqk, Pcq, Pck, Vtq, Vtc, DE, out);
}
